// JointRetention_72533407695347
// MI455X (gfx1250) — matrix operand order repaired and checked by reading, not compiled, not run on hardware
//
#include <hip/hip_runtime.h>
#include <hip/hip_bf16.h>

typedef __bf16 bf16_t;
typedef __attribute__((ext_vector_type(16))) __bf16 v16bf;
typedef __attribute__((ext_vector_type(8)))  __bf16 v8bf;
typedef __attribute__((ext_vector_type(4)))  __bf16 v4bf;
typedef __attribute__((ext_vector_type(8)))  float  v8f;
typedef __attribute__((ext_vector_type(4)))  float  f32x4;
typedef __attribute__((ext_vector_type(4)))  unsigned int u32x4;
typedef __attribute__((ext_vector_type(8)))  int    i32x8;
typedef __attribute__((ext_vector_type(4)))  int    i32x4;

#define SEQL   243
#define HID    256
#define CHUNKL 81
#define NJ     17
#define BNJ    1088   // BATCH * NUM_JOINTS

#ifndef __has_builtin
#define __has_builtin(x) 0
#endif
#if __has_builtin(__builtin_amdgcn_tensor_load_to_lds)
#define HAVE_TDM 1
#else
#define HAVE_TDM 0
#endif

// Assemble a 16-element bf16 WMMA fragment from two aligned 16-byte chunks.
__device__ __forceinline__ v16bf make_frag(const bf16_t* lo, const bf16_t* hi) {
  v8bf a = *(const v8bf*)lo;
  v8bf b = *(const v8bf*)hi;
  v16bf r;
#pragma unroll
  for (int i = 0; i < 8; ++i) { r[i] = a[i]; r[i + 8] = b[i]; }
  return r;
}

__device__ __forceinline__ v8f wmma_bf16(v16bf a, v16bf b, v8f c) {
  return __builtin_amdgcn_wmma_f32_16x16x32_bf16(false, a, false, b, (short)0, c,
                                                 false, false);
}

__device__ __forceinline__ void wait_tensorcnt0() {
#if __has_builtin(__builtin_amdgcn_s_wait_tensorcnt)
  __builtin_amdgcn_s_wait_tensorcnt(0);
#else
  asm volatile("s_wait_tensorcnt 0x0" ::: "memory");
#endif
}

#if HAVE_TDM
// Issue a 2D TDM tile load (global -> LDS). All args wave-uniform.
// Units: data_size-sized elements (we use 4-byte elements everywhere).
// OOB rows (beyond tensor_d1) are hardware zero-filled in LDS.
__device__ __forceinline__ void tdm_load_2d(unsigned lds_addr, const void* gaddr,
                                            unsigned tensor_d0, unsigned tensor_d1,
                                            unsigned tile_d0, unsigned tile_d1,
                                            unsigned stride0) {
  unsigned long long ga = (unsigned long long)gaddr;
  // D# group 0: count=1 | lds_addr[63:32] | global_addr[120:64] | type=2 (bits 127:126)
  u32x4 g0;
  g0[0] = 1u;
  g0[1] = lds_addr;
  g0[2] = (unsigned)ga;
  g0[3] = (unsigned)((ga >> 32) & 0x01FFFFFFu) | (2u << 30);
  // D# group 1: wg_mask=0 | data_size=2 (4B) | dims/strides per ISA 8.4
  i32x8 g1;
  g1[0] = (int)(2u << 16);                                         // data_size=4B
  g1[1] = (int)((tensor_d0 & 0xFFFFu) << 16);                      // tensor_dim0[15:0] @63:48
  g1[2] = (int)((tensor_d0 >> 16) | ((tensor_d1 & 0xFFFFu) << 16));// td0[31:16] | td1[15:0]
  g1[3] = (int)((tensor_d1 >> 16) | (tile_d0 << 16));              // td1[31:16] | tile_dim0
  g1[4] = (int)tile_d1;                                            // tile_dim1 | tile_dim2=0
  g1[5] = (int)stride0;                                            // tensor_dim0_stride[31:0]
  g1[6] = 0;                                                       // stride0[47:32]|stride1 lo
  g1[7] = 0;
  i32x4 z4 = {0, 0, 0, 0};
#if __clang_major__ >= 23
  i32x8 z8 = {0, 0, 0, 0, 0, 0, 0, 0};
  __builtin_amdgcn_tensor_load_to_lds(g0, g1, z4, z4, z8, 0);
#else
  __builtin_amdgcn_tensor_load_to_lds(g0, g1, z4, z4, 0);
#endif
}
#endif

// ---------------------------------------------------------------------------
// Kernel 0: W (256x256 f32, k-major) -> WT bf16 [mat][n][k].
// ---------------------------------------------------------------------------
__global__ __launch_bounds__(256) void wt_prep(const float* __restrict__ WQ,
                                               const float* __restrict__ WK,
                                               const float* __restrict__ WV,
                                               bf16_t* __restrict__ wt) {
  int mat = blockIdx.x >> 8;
  int n   = blockIdx.x & 255;
  int k   = threadIdx.x;
  const float* W = (mat == 0) ? WQ : ((mat == 1) ? WK : WV);
  wt[((size_t)mat * 256 + n) * 256 + k] = (bf16_t)W[(size_t)k * 256 + n];
}

// ---------------------------------------------------------------------------
// Kernel 1: Q/K/V projections via bf16 WMMA + xpos epilogue.
// ---------------------------------------------------------------------------
__global__ __launch_bounds__(256) void proj_xpos(const float* __restrict__ X,
                                                 const bf16_t* __restrict__ wt,
                                                 bf16_t* __restrict__ qws,
                                                 bf16_t* __restrict__ kws,
                                                 bf16_t* __restrict__ vtws) {
  __shared__ bf16_t xl[64 * 256];           // 32 KB X tile (bf16)

  const int b    = blockIdx.y;
  const int rb   = blockIdx.x * 64;
  const int tid  = threadIdx.x;
  const int wave = tid >> 5;
  const int lane = tid & 31;
  const int hf   = lane >> 4;
  const int l16  = lane & 15;

  const float* Xb = X + (size_t)b * SEQL * HID;
#pragma unroll
  for (int it = 0; it < 16; ++it) {
    int e   = (it * 256 + tid) * 4;
    int row = e >> 8, col = e & 255;
    f32x4 v = {};
    if (rb + row < SEQL) v = *(const f32x4*)(Xb + (size_t)(rb + row) * HID + col);
    v4bf o = {(bf16_t)v[0], (bf16_t)v[1], (bf16_t)v[2], (bf16_t)v[3]};
    *(v4bf*)&xl[row * 256 + col] = o;
  }
  __syncthreads();

  const float L2_10000 = 13.28771238f;      // log2(10000)

  for (int mat = 0; mat < 3; ++mat) {
    const bf16_t* wtm = wt + (size_t)mat * 256 * 256;
#pragma unroll 1
    for (int mt = 0; mt < 4; ++mt) {
      v8f acc0 = {0.f, 0.f, 0.f, 0.f, 0.f, 0.f, 0.f, 0.f};
      v8f acc1 = acc0;
#pragma unroll
      for (int dc = 0; dc < 8; ++dc) {
        const bf16_t* ar = &xl[(mt * 16 + l16) * 256 + dc * 32 + 8 * hf];
        v16bf A = make_frag(ar, ar + 16);
        const bf16_t* b0 = &wtm[(size_t)(wave * 32 + l16) * 256 + dc * 32 + 8 * hf];
        v16bf B0 = make_frag(b0, b0 + 16);
        acc0 = wmma_bf16(A, B0, acc0);
        const bf16_t* b1 = b0 + 16 * 256;
        v16bf B1 = make_frag(b1, b1 + 16);
        acc1 = wmma_bf16(A, B1, acc1);
      }
#pragma unroll
      for (int nt = 0; nt < 2; ++nt) {
        v8f acc = nt ? acc1 : acc0;
        int n = wave * 32 + nt * 16 + l16;
        if (mat == 2) {
#pragma unroll
          for (int r = 0; r < 8; ++r) {
            int l = rb + mt * 16 + r + 8 * hf;
            if (l < SEQL) vtws[((size_t)b * 256 + n) * 256 + l] = (bf16_t)acc[r];
          }
        } else {
          int   c     = n >> 1;
          float invf  = exp2f((float)c * (-L2_10000 / 128.f));
          float logbs = log2f(((float)(2 * c) + 102.4f) * (1.0f / 358.4f));
          float sgn   = (n & 1) ? 1.f : -1.f;
          float sdir  = (mat == 0) ? (1.f / 512.f) : (-1.f / 512.f);
          bf16_t* dst = (mat == 0) ? qws : kws;
#pragma unroll
          for (int r = 0; r < 8; ++r) {
            float a = acc[r];
            float p = __shfl_xor(a, 1, 32);
            int   l = rb + mt * 16 + r + 8 * hf;
            float lf = (float)l;
            float sc = exp2f(logbs * lf * sdir);
            float th = lf * invf;
            float cs = __cosf(th) * sc;
            float ss = __sinf(th) * sc;
            float o  = a * cs + sgn * p * ss;
            if (l < SEQL) dst[((size_t)b * SEQL + l) * 256 + n] = (bf16_t)o;
          }
        }
      }
    }
  }
}

// ---------------------------------------------------------------------------
// Kernel 2: retention. K/V^T blocks staged into LDS by the Tensor Data Mover
// (TDM) when available; WMMA for S = Q K^T and O += (S .* D) V.
// ---------------------------------------------------------------------------
__global__ __launch_bounds__(256) void retention(const bf16_t* __restrict__ qws,
                                                 const bf16_t* __restrict__ kws,
                                                 const bf16_t* __restrict__ vtws,
                                                 const float* __restrict__ gamma,
                                                 float* __restrict__ out) {
  __shared__ bf16_t qlds[64 * 256];   // 32 KB Q tile
  __shared__ bf16_t klds[32 * 256];   // 16 KB K block
  __shared__ bf16_t vtlds[256 * 32];  // 16 KB V^T block
  __shared__ bf16_t slds[64 * 32];    //  4 KB decayed S block

  const int b    = blockIdx.y;
  const int rb   = blockIdx.x * 64;
  const int tid  = threadIdx.x;
  const int wave = tid >> 5;
  const int lane = tid & 31;
  const int hf   = lane >> 4;
  const int l16  = lane & 15;

  const float log2g = log2f(gamma[b % NJ]);

  // Load Q tile 64x256 bf16 (zero-pad rows >= 243).
  const bf16_t* qb = qws + (size_t)b * SEQL * 256;
#pragma unroll
  for (int it = 0; it < 8; ++it) {
    int c    = it * 256 + tid;
    int row  = c >> 5;
    int col8 = (c & 31) * 8;
    u32x4 v = {};
    if (rb + row < SEQL) v = *(const u32x4*)(qb + (size_t)(rb + row) * 256 + col8);
    *(u32x4*)&qlds[row * 256 + col8] = v;
  }

  const int sm = wave >> 1, sn = wave & 1;   // phase-A tile (S = 64x32)
  const int om = wave & 3, og = wave >> 2;   // phase-B tile (O = 64x256)

  v8f oacc[8];
  {
    v8f z = {0.f, 0.f, 0.f, 0.f, 0.f, 0.f, 0.f, 0.f};
#pragma unroll
    for (int i = 0; i < 8; ++i) oacc[i] = z;
  }

  const bf16_t* kbp = kws + (size_t)b * SEQL * 256;
  const bf16_t* vbp = vtws + (size_t)b * 256 * 256;

  for (int kb = 0; kb < 256; kb += 32) {
    __syncthreads();  // previous iteration done with K/V/S LDS

#if HAVE_TDM
    // TDM tile loads (4-byte elements). K: 32 rows x 128 f32 (=512B) with
    // hardware zero-fill past row (SEQL-kb). V^T: 256 rows x 16 f32 (=64B)
    // from column kb, global row stride 512B, packed contiguously into LDS.
    if (wave == 0) {
      tdm_load_2d((unsigned)(size_t)&klds[0], kbp + (size_t)kb * 256,
                  /*tensor_d0=*/128u, /*tensor_d1=*/(unsigned)(SEQL - kb),
                  /*tile_d0=*/128u, /*tile_d1=*/32u, /*stride0=*/128u);
    } else if (wave == 1) {
      tdm_load_2d((unsigned)(size_t)&vtlds[0], vbp + kb,
                  /*tensor_d0=*/(unsigned)(128 - kb / 2), /*tensor_d1=*/256u,
                  /*tile_d0=*/16u, /*tile_d1=*/256u, /*stride0=*/128u);
    }
    if (wave < 2) wait_tensorcnt0();
#else
    // Fallback: cooperative vector loads.
#pragma unroll
    for (int it = 0; it < 4; ++it) {
      int c    = it * 256 + tid;
      int row  = c >> 5;
      int col8 = (c & 31) * 8;
      u32x4 v = {};
      if (kb + row < SEQL) v = *(const u32x4*)(kbp + (size_t)(kb + row) * 256 + col8);
      *(u32x4*)&klds[row * 256 + col8] = v;
    }
#pragma unroll
    for (int it = 0; it < 4; ++it) {
      int c = it * 256 + tid;
      int d = c >> 2;
      int q = (c & 3) * 8;
      u32x4 v = *(const u32x4*)(vbp + (size_t)d * 256 + kb + q);
      *(u32x4*)&vtlds[d * 32 + q] = v;
    }
#endif
    // Prefetch next K block into caches (global_prefetch_b8).
    if (kb + 32 < SEQL)
      __builtin_prefetch(kbp + (size_t)(kb + 32 + (tid >> 3)) * 256 + (tid & 7) * 32, 0, 1);

    __syncthreads();

    // ---- Phase A: S tile (16x16) at (sm, sn) over d = 256 ----
    v8f sacc = {0.f, 0.f, 0.f, 0.f, 0.f, 0.f, 0.f, 0.f};
#pragma unroll
    for (int dc = 0; dc < 8; ++dc) {
      const bf16_t* ar = &qlds[(sm * 16 + l16) * 256 + dc * 32 + 8 * hf];
      v16bf A = make_frag(ar, ar + 16);
      const bf16_t* br = &klds[(sn * 16 + l16) * 256 + dc * 32 + 8 * hf];
      v16bf Bm = make_frag(br, br + 16);
      sacc = wmma_bf16(A, Bm, sacc);
    }
    // Decay mask in C-layout: element r -> (i = query pos, j = key pos).
    {
      int j = kb + sn * 16 + l16;
#pragma unroll
      for (int r = 0; r < 8; ++r) {
        int m   = r + 8 * hf;
        int i   = rb + sm * 16 + m;
        int lim = (i / CHUNKL + 1) * CHUNKL;
        float d = (j < lim) ? exp2f(log2g * fabsf((float)(i - j))) : 0.f;
        slds[(sm * 16 + m) * 32 + sn * 16 + l16] = (bf16_t)(sacc[r] * d);
      }
    }
    __syncthreads();

    // ---- Phase B: O(16x128 per wave) += S(16x32) * V(32x128) ----
    const bf16_t* ar = &slds[(om * 16 + l16) * 32 + 8 * hf];
    v16bf A = make_frag(ar, ar + 16);
#pragma unroll
    for (int nt = 0; nt < 8; ++nt) {
      const bf16_t* br = &vtlds[(og * 128 + nt * 16 + l16) * 32 + 8 * hf];
      v16bf Bm = make_frag(br, br + 16);
      oacc[nt] = wmma_bf16(A, Bm, oacc[nt]);
    }
  }

  // Store O (fp32).
  float* ob = out + (size_t)b * SEQL * 256;
#pragma unroll
  for (int nt = 0; nt < 8; ++nt) {
#pragma unroll
    for (int r = 0; r < 8; ++r) {
      int i = rb + om * 16 + r + 8 * hf;
      if (i < SEQL) ob[(size_t)i * 256 + og * 128 + nt * 16 + l16] = oacc[nt][r];
    }
  }
}

// ---------------------------------------------------------------------------
extern "C" void kernel_launch(void* const* d_in, const int* in_sizes, int n_in,
                              void* d_out, int out_size, void* d_ws, size_t ws_size,
                              hipStream_t stream) {
  const float* X     = (const float*)d_in[0];
  const float* WQ    = (const float*)d_in[1];
  const float* WK    = (const float*)d_in[2];
  const float* WV    = (const float*)d_in[3];
  const float* gamma = (const float*)d_in[4];
  float* out = (float*)d_out;

  char* ws = (char*)d_ws;
  const size_t wtBytes = 3ull * 256 * 256 * sizeof(bf16_t);
  const size_t qkBytes = (size_t)BNJ * SEQL * 256 * sizeof(bf16_t);
  bf16_t* wt   = (bf16_t*)ws;
  bf16_t* qws  = (bf16_t*)(ws + wtBytes);
  bf16_t* kws  = (bf16_t*)(ws + wtBytes + qkBytes);
  bf16_t* vtws = (bf16_t*)(ws + wtBytes + 2 * qkBytes);

  wt_prep<<<dim3(3 * 256), dim3(256), 0, stream>>>(WQ, WK, WV, wt);
  proj_xpos<<<dim3(4, BNJ), dim3(256), 0, stream>>>(X, wt, qws, kws, vtws);
  retention<<<dim3(4, BNJ), dim3(256), 0, stream>>>(qws, kws, vtws, gamma, out);
}
